// GlobalAttentionPooling_44401371906310
// MI455X (gfx1250) — matrix operand order repaired and checked by reading, not compiled, not run on hardware
//
#include <hip/hip_runtime.h>
#include <hip/hip_bf16.h>

// ---------------------------------------------------------------------------
// GlobalAttentionPooling for MI455X (gfx1250, wave32, WMMA)
//   scores = tanh(x@W1+b1)@W2+b2 ; w = softmax(scores over all N)
//   out[b] = (sum_{i in b} x_i * w_i) / max(count_b,1)
// Softmax max-pass is skipped: |score| <= ||W2||_1 + |b2| ~ 12, so exp() is
// safely bounded in f32 (Z <= 5e5 * e^12 << FLT_MAX).
// ---------------------------------------------------------------------------

#define D_DIM 256   // hidden dim of x
#define H_DIM 128   // attention MLP hidden

typedef __attribute__((ext_vector_type(16))) __bf16 v16bf;
typedef __attribute__((ext_vector_type(8)))  float  v8f;

// f32 pair -> packed bf16 dword. Round-half-up then grab the two high
// halfwords with a single v_perm_b32 (selector 0x07060302: bytes {S0.b3,
// S0.b2, S1.b3, S1.b2} -> {hi[31:16], lo[31:16]}).
__device__ __forceinline__ unsigned pack2(float lo, float hi) {
  unsigned ul = __float_as_uint(lo) + 0x8000u;
  unsigned uh = __float_as_uint(hi) + 0x8000u;
  return __builtin_amdgcn_perm(uh, ul, 0x07060302u);
}

// gfx1250 has V_TANH_F32 as a TRANS32 op (co-executes with VALU).
#if __has_builtin(__builtin_amdgcn_tanhf)
__device__ __forceinline__ float fast_tanh(float x) {
  return __builtin_amdgcn_tanhf(x);
}
#else
__device__ __forceinline__ float fast_tanh(float x) { return tanhf(x); }
#endif

// ---------------------------------------------------------------------------
// K0: zero accumulators (acc[B*D], counts[B], Z) -- contiguous region
// ---------------------------------------------------------------------------
__global__ void k0_zero(float* __restrict__ p, int n) {
  int i = blockIdx.x * 256 + threadIdx.x;
  if (i < n) p[i] = 0.0f;
}

// ---------------------------------------------------------------------------
// K1: attention MLP via bf16 WMMA. 256 threads = 8 waves, each wave owns a
// 16-row tile -> 128 rows / block. W1 staged in LDS as bf16, XOR-swizzled in
// 16B units so B-fragment ds_load_b128's are bank-conflict free.
// ---------------------------------------------------------------------------
__global__ __launch_bounds__(256) void k1_scores(
    const float* __restrict__ x, const float* __restrict__ W1,
    const float* __restrict__ b1, const float* __restrict__ W2,
    const float* __restrict__ b2, float* __restrict__ eBuf,
    float* __restrict__ Z, int nNodes) {
  // LDS: W1 transposed to [n][k] bf16, stored as 16B units (8 k-values).
  // unit index = n*32 + (kg ^ (n&15))  -> exactly 64 KB.
  __shared__ uint4 w1q[H_DIM * 32];

  const int tid = threadIdx.x;

  // ---- cooperative W1 fill: logical unit (n, kg) covers k = kg*8 .. kg*8+7
  for (int idx = tid; idx < H_DIM * 32; idx += 256) {
    int n = idx & (H_DIM - 1);
    int kg = idx >> 7;
    float f[8];
#pragma unroll
    for (int j = 0; j < 8; ++j) f[j] = W1[(kg * 8 + j) * H_DIM + n];
    uint4 q;
    q.x = pack2(f[0], f[1]); q.y = pack2(f[2], f[3]);
    q.z = pack2(f[4], f[5]); q.w = pack2(f[6], f[7]);
    w1q[n * 32 + (kg ^ (n & 15))] = q;
  }
  __syncthreads();

  const int lane = tid & 31;
  const int wv   = tid >> 5;
  const int m    = lane & 15;   // A-row within tile / C-column / B-column
  const int hi   = lane >> 4;   // half-wave select
  const int rowBase = blockIdx.x * 128 + wv * 16;
  const int arow = min(rowBase + m, nNodes - 1);   // clamp (EXEC must stay full)
  const float4* xr = (const float4*)(x + (size_t)arow * D_DIM);

  float b1v[8], w2v[8];
#pragma unroll
  for (int nt = 0; nt < 8; ++nt) {
    b1v[nt] = b1[nt * 16 + m];
    w2v[nt] = W2[nt * 16 + m];
  }
  const float b2s = b2[0];

  v8f acc[8];
#pragma unroll
  for (int nt = 0; nt < 8; ++nt)
#pragma unroll
    for (int r = 0; r < 8; ++r) acc[nt][r] = 0.0f;

  // ---- main GEMM: K = 256 = 8 chunks of 32; N = 128 = 8 tiles of 16
#pragma unroll
  for (int kc = 0; kc < 8; ++kc) {
    if (kc < 7) __builtin_prefetch((const void*)(xr + kc * 8 + 8), 0, 1);
    // A fragment (16x32 bf16): lane holds row m, K = kc*32 + hi*8 + {0..7}
    // in VGPR0-3 and K = kc*32 + 16 + hi*8 + {0..7} in VGPR4-7.
    float4 f0 = xr[kc * 8 + hi * 2];
    float4 f1 = xr[kc * 8 + hi * 2 + 1];
    float4 f2 = xr[kc * 8 + 4 + hi * 2];
    float4 f3 = xr[kc * 8 + 4 + hi * 2 + 1];
    union { uint4 q[2]; v16bf v; } af;
    af.q[0] = make_uint4(pack2(f0.x, f0.y), pack2(f0.z, f0.w),
                         pack2(f1.x, f1.y), pack2(f1.z, f1.w));
    af.q[1] = make_uint4(pack2(f2.x, f2.y), pack2(f2.z, f2.w),
                         pack2(f3.x, f3.y), pack2(f3.z, f3.w));
#pragma unroll
    for (int nt = 0; nt < 8; ++nt) {
      // B fragment (32x16 bf16): lane holds column n = nt*16+m,
      // K = kc*32 + hi*8 + {0..7}, kc*32 + 16 + hi*8 + {0..7}  -> two swizzled 16B LDS units.
      const int n = nt * 16 + m;
      const int kg0 = kc * 4 + hi;
      union { uint4 q[2]; v16bf v; } bf;
      bf.q[0] = w1q[n * 32 + ((kg0    ) ^ (n & 15))];
      bf.q[1] = w1q[n * 32 + ((kg0 + 2) ^ (n & 15))];
      acc[nt] = __builtin_amdgcn_wmma_f32_16x16x32_bf16(
          false, af.v, false, bf.v, (short)0, acc[nt], false, false);
    }
  }

  // ---- epilogue: s = sum_n tanh(h+b1[n])*W2[n] + b2 ; e = exp(s)
  // C layout: acc[nt][r] is (row = r + 8*hi, col n = nt*16 + m).
  float part[8];
#pragma unroll
  for (int r = 0; r < 8; ++r) part[r] = 0.0f;
#pragma unroll
  for (int nt = 0; nt < 8; ++nt) {
    const float bn = b1v[nt], wn = w2v[nt];
#pragma unroll
    for (int r = 0; r < 8; ++r) part[r] += fast_tanh(acc[nt][r] + bn) * wn;
  }
  // reduce the 128 columns: butterfly across the 16-lane half-wave
#pragma unroll
  for (int msk = 1; msk < 16; msk <<= 1)
#pragma unroll
    for (int r = 0; r < 8; ++r) part[r] += __shfl_xor(part[r], msk, 32);

  if (m == 0) {
    float zsum = 0.0f;
#pragma unroll
    for (int r = 0; r < 8; ++r) {
      int grow = rowBase + r + 8 * hi;
      if (grow < nNodes) {
        float ev = __expf(part[r] + b2s);   // bounded, no max-shift needed
        eBuf[grow] = ev;
        zsum += ev;
      }
    }
    atomicAdd(Z, zsum);
  }
}

// ---------------------------------------------------------------------------
// K2: weighted segment sums. One block = 256 contiguous nodes, one thread =
// one of the 256 columns (fully coalesced). batch is sorted -> register
// accumulation, atomicAdd only on segment boundaries.
// ---------------------------------------------------------------------------
__global__ __launch_bounds__(256) void k2_segsum(
    const float* __restrict__ x, const int* __restrict__ batch,
    const float* __restrict__ e, float* __restrict__ acc,
    float* __restrict__ counts, int nNodes) {
  __shared__ int   sb[256];
  __shared__ float se[256];
  const int i0 = blockIdx.x * 256;
  if (i0 >= nNodes) return;
  const int tid = threadIdx.x;
  const int i = i0 + tid;
  if (i < nNodes) { sb[tid] = batch[i]; se[tid] = e[i]; }
  __syncthreads();

  const int nloc = min(256, nNodes - i0);
  int cur = sb[0];
  float a = 0.0f, c = 0.0f;
  for (int j = 0; j < nloc; ++j) {
    int b = sb[j];
    if (b != cur) {                       // segment boundary: flush
      atomicAdd(&acc[(size_t)cur * D_DIM + tid], a);
      if (tid == 0) atomicAdd(&counts[cur], c);
      a = 0.0f; c = 0.0f; cur = b;
    }
    a += x[(size_t)(i0 + j) * D_DIM + tid] * se[j];
    c += 1.0f;
    if (j + 4 < nloc)
      __builtin_prefetch(&x[(size_t)(i0 + j + 4) * D_DIM + tid], 0, 1);
  }
  atomicAdd(&acc[(size_t)cur * D_DIM + tid], a);
  if (tid == 0) atomicAdd(&counts[cur], c);
}

// ---------------------------------------------------------------------------
// K3: out[b,d] = acc[b,d] / (Z * max(count_b, 1))
// ---------------------------------------------------------------------------
__global__ __launch_bounds__(256) void k3_norm(
    const float* __restrict__ acc, const float* __restrict__ counts,
    const float* __restrict__ Z, float* __restrict__ out, int Bseg) {
  const int b = blockIdx.x;
  const int d = threadIdx.x;
  if (b >= Bseg) return;
  const float inv = 1.0f / ((*Z) * fmaxf(counts[b], 1.0f));
  out[(size_t)b * D_DIM + d] = acc[(size_t)b * D_DIM + d] * inv;
}

// ---------------------------------------------------------------------------
extern "C" void kernel_launch(void* const* d_in, const int* in_sizes, int n_in,
                              void* d_out, int out_size, void* d_ws, size_t ws_size,
                              hipStream_t stream) {
  const float* x     = (const float*)d_in[0];
  const int*   batch = (const int*)d_in[1];   // sorted segment ids
  const float* W1    = (const float*)d_in[2];
  const float* b1    = (const float*)d_in[3];
  const float* W2    = (const float*)d_in[4];
  const float* b2    = (const float*)d_in[5];

  const int N    = in_sizes[0] / D_DIM;   // num nodes
  const int Bseg = out_size / D_DIM;      // num segments

  // workspace layout (floats): e[N] | acc[B*D] | counts[B] | Z[1]
  float* ws     = (float*)d_ws;
  float* eBuf   = ws;
  float* accBuf = ws + N;
  float* cntBuf = accBuf + (size_t)Bseg * D_DIM;
  float* zBuf   = cntBuf + Bseg;

  const int nzero = Bseg * D_DIM + Bseg + 1;   // acc+counts+Z are contiguous
  k0_zero<<<(nzero + 255) / 256, 256, 0, stream>>>(accBuf, nzero);

  k1_scores<<<(N + 127) / 128, 256, 0, stream>>>(x, W1, b1, W2, b2,
                                                 eBuf, zBuf, N);

  k2_segsum<<<(N + 255) / 256, 256, 0, stream>>>(x, batch, eBuf,
                                                 accBuf, cntBuf, N);

  k3_norm<<<Bseg, 256, 0, stream>>>(accBuf, cntBuf, zBuf, (float*)d_out, Bseg);
}
